// CapsuleBlock_70695161692767
// MI455X (gfx1250) — matrix operand order repaired and checked by reading, not compiled, not run on hardware
//
#include <hip/hip_runtime.h>

typedef __attribute__((ext_vector_type(16))) _Float16 v16h;
typedef __attribute__((ext_vector_type(8)))  _Float16 v8h;
typedef __attribute__((ext_vector_type(8)))  float    v8f;

#define BB 2
#define PP 8192
#define NVOX 24
#define NVOX3 (24*24*24)

// ---------------------------------------------------------------- utilities
__global__ void zero_f32(float* p, long cnt) {
  long i = (long)blockIdx.x * blockDim.x + threadIdx.x;
  long st = (long)gridDim.x * blockDim.x;
  for (; i < cnt; i += st) p[i] = 0.f;
}

__global__ void cvt_f16_k(const float* __restrict__ in, _Float16* __restrict__ out, long cnt) {
  long i = (long)blockIdx.x * blockDim.x + threadIdx.x;
  long st = (long)gridDim.x * blockDim.x;
  for (; i < cnt; i += st) out[i] = (_Float16)in[i];
}

// per-cloud axis-aligned bounding box (min/max over P points, 3 axes)
__global__ void minmax_k(const float* __restrict__ pcl, float* __restrict__ mnmx, int Pn) {
  int b = blockIdx.x;
  int tid = threadIdx.x;
  float mn[3] = {1e30f, 1e30f, 1e30f}, mx[3] = {-1e30f, -1e30f, -1e30f};
  for (int p = tid; p < Pn; p += blockDim.x) {
    const float* q = pcl + ((long)b * Pn + p) * 3;
    for (int a = 0; a < 3; ++a) {
      float v = q[a];
      mn[a] = fminf(mn[a], v);
      mx[a] = fmaxf(mx[a], v);
    }
  }
  __shared__ float smn[256][3], smx[256][3];
  for (int a = 0; a < 3; ++a) { smn[tid][a] = mn[a]; smx[tid][a] = mx[a]; }
  __syncthreads();
  for (int s = 128; s > 0; s >>= 1) {
    if (tid < s)
      for (int a = 0; a < 3; ++a) {
        smn[tid][a] = fminf(smn[tid][a], smn[tid + s][a]);
        smx[tid][a] = fmaxf(smx[tid][a], smx[tid + s][a]);
      }
    __syncthreads();
  }
  if (tid == 0)
    for (int a = 0; a < 3; ++a) {
      mnmx[b * 6 + a]     = smn[0][a];
      mnmx[b * 6 + 3 + a] = smx[0][a];
    }
}

// voxelize + scatter-add point features into dense (B,128,24^3) grid; record vid
__global__ void scatter_k(const float* __restrict__ pcl, const float* __restrict__ feat,
                          const float* __restrict__ mnmx, float* __restrict__ mesh,
                          int* __restrict__ vid, int Pn, int Cf) {
  int bp = blockIdx.x;
  int b = bp / Pn, p = bp % Pn;
  const float* q = pcl + ((long)b * Pn + p) * 3;
  int id3[3];
  for (int a = 0; a < 3; ++a) {
    float mn = mnmx[b * 6 + a], mx = mnmx[b * 6 + 3 + a];
    int ix = (int)floorf((q[a] - mn) / (mx - mn + 1e-9f) * (float)NVOX);
    id3[a] = min(max(ix, 0), NVOX - 1);
  }
  int v = id3[0] * NVOX * NVOX + id3[1] * NVOX + id3[2];
  int c = threadIdx.x;
  if (c == 0) vid[(long)b * Pn + p] = v;
  atomicAdd(&mesh[((long)b * Cf + c) * NVOX3 + v], feat[((long)b * Pn + p) * Cf + c]);
}

// ---------------- weight prep: f32 -> f16, K padded to /32, Cout padded rows
__global__ void prep_wdir_f16(const float* __restrict__ wi, _Float16* __restrict__ wo,
                              int Cout, int Ktot, int KtotP, long total) {
  long idx = (long)blockIdx.x * blockDim.x + threadIdx.x;
  if (idx >= total) return;
  int kk = (int)(idx % KtotP);
  long n = idx / KtotP;
  float v = 0.f;
  if (n < Cout && kk < Ktot) v = wi[n * Ktot + kk];
  wo[idx] = (_Float16)v;
}

// deconv weights: flip(k) + transpose (Cin,Cout,27) -> padded f16 [co][ci*27+kf]
__global__ void prep_wdec_f16(const float* __restrict__ wi, _Float16* __restrict__ wo,
                              int Cin, int Cout, int KtotP, long total) {
  long idx = (long)blockIdx.x * blockDim.x + threadIdx.x;
  if (idx >= total) return;
  int kk = (int)(idx % KtotP);
  long co = idx / KtotP;
  int Ktot = Cin * 27;
  float v = 0.f;
  if (co < Cout && kk < Ktot) {
    int ci = kk / 27; int kf = kk - ci * 27;
    int kd = kf / 9;  int r2 = kf - kd * 9;
    int kh = r2 / 3;  int kw = r2 - kh * 3;
    v = wi[((long)ci * Cout + co) * 27 + (2 - kd) * 9 + (2 - kh) * 3 + (2 - kw)];
  }
  wo[idx] = (_Float16)v;
}

// per-layer K-index -> input gather offset table (independent of output position)
__global__ void prep_offs_k(int* __restrict__ offs, int Cin, int KS,
                            int IHW, int IW, int IDHW, int Ktot, int KtotP) {
  int idx = blockIdx.x * blockDim.x + threadIdx.x;
  if (idx >= KtotP) return;
  int v = 0;
  if (idx < Ktot) {
    int K3 = KS * KS * KS;
    int ci = idx / K3;      int r  = idx - ci * K3;
    int kd = r / (KS * KS); int r2 = r - kd * KS * KS;
    int kh = r2 / KS;       int kw = r2 - kh * KS;
    v = ci * IDHW + kd * IHW + kh * IW + kw;
  }
  offs[idx] = v;
}

// ------------------------------------------- implicit-GEMM conv3d via WMMA
// Block: 256 threads = 8 wave32 (2 M-halves x 4 N-quarters), tile M=64 x N=256.
// K-step 32, f16 WMMA / f32 acc, double-buffered LDS, B tile staged with
// gfx1250 global_load_async_to_lds_b128 (ASYNCcnt), A tile via offset table.
template <bool RELU, bool OUTF16, bool SPLITK>
__global__ void __launch_bounds__(256) conv3d_wmma(
    const _Float16* __restrict__ in, const _Float16* __restrict__ w,
    const float* __restrict__ bias, const int* __restrict__ offs,
    void* __restrict__ out,
    int Cin, int Cout, int stride,
    int ID, int IH, int IW, int OD, int OH, int OW, int Ktot) {
  const int tid = threadIdx.x;
  const int Mbase = blockIdx.x * 64;
  const int Nbase = blockIdx.y * 256;
  const int OHW = OH * OW, Msz = OD * OHW;
  const int IHW = IH * IW, IDHW = ID * IHW;
  const int KtotP = (Ktot + 31) & ~31;
  const int nkTot = KtotP >> 5;
  const int nkLoc = nkTot / gridDim.z;
  const int kt0 = blockIdx.z * nkLoc;

  __shared__ __align__(16) _Float16 sA[2][64][32];
  __shared__ __align__(16) _Float16 sB[2][256][32];

  // A staging: thread -> rows (amRow, amRow+32), k-group of 4
  const int amRow = tid & 31;
  const int akq = tid >> 5;
  const _Float16* inb0;
  const _Float16* inb1;
  {
    int m0 = Mbase + amRow;
    int b_ = m0 / Msz;  int ms = m0 - b_ * Msz;
    int od = ms / OHW;  int mr = ms - od * OHW;
    int oh = mr / OW;   int ow = mr - oh * OW;
    inb0 = in + (long)b_ * Cin * IDHW + (long)(od * stride) * IHW + (oh * stride) * IW + (ow * stride);
    int m1 = m0 + 32;
    b_ = m1 / Msz;  ms = m1 - b_ * Msz;
    od = ms / OHW;  mr = ms - od * OHW;
    oh = mr / OW;   ow = mr - oh * OW;
    inb1 = in + (long)b_ * Cin * IDHW + (long)(od * stride) * IHW + (oh * stride) * IW + (ow * stride);
  }
  const _Float16* wrow = w + (long)(Nbase + tid) * KtotP;

  const int wid = tid >> 5, lane = tid & 31;
  const int wm = wid & 1, wn = wid >> 1;
  const int half = lane >> 4, lrow = lane & 15;

  v8f acc[2][4] = {};

  auto stageA = [&](int kt, uint2& u0, uint2& u1) {
    int kb = kt << 5;
    int4 o4 = *(const int4*)(offs + kb + akq * 4);  // same for all m
    int kk = kb + akq * 4;
    _Float16 z = (_Float16)0.f;
    _Float16 a0 = inb0[o4.x], a1 = inb0[o4.y], a2 = inb0[o4.z], a3 = inb0[o4.w];
    _Float16 b0 = inb1[o4.x], b1 = inb1[o4.y], b2 = inb1[o4.z], b3 = inb1[o4.w];
    a0 = (kk + 0 < Ktot) ? a0 : z;  b0 = (kk + 0 < Ktot) ? b0 : z;
    a1 = (kk + 1 < Ktot) ? a1 : z;  b1 = (kk + 1 < Ktot) ? b1 : z;
    a2 = (kk + 2 < Ktot) ? a2 : z;  b2 = (kk + 2 < Ktot) ? b2 : z;
    a3 = (kk + 3 < Ktot) ? a3 : z;  b3 = (kk + 3 < Ktot) ? b3 : z;
    union { _Float16 h[4]; uint2 u; } p0, p1;
    p0.h[0] = a0; p0.h[1] = a1; p0.h[2] = a2; p0.h[3] = a3;
    p1.h[0] = b0; p1.h[1] = b1; p1.h[2] = b2; p1.h[3] = b3;
    u0 = p0.u; u1 = p1.u;
  };
  auto asyncB = [&](int kt, int buf) {
    const _Float16* src = wrow + (kt << 5);  // 64 B per thread
    unsigned dst = (unsigned)(size_t)&sB[buf][tid][0];
#pragma unroll
    for (int j = 0; j < 4; ++j)
      asm volatile("global_load_async_to_lds_b128 %0, %1, off"
                   :: "v"(dst + 16u * j), "v"(src + 8 * j) : "memory");
  };

  // prologue: fill buffer 0
  {
    uint2 u0, u1;
    stageA(kt0, u0, u1);
    asyncB(kt0, 0);
    *(uint2*)&sA[0][amRow][akq * 4] = u0;
    *(uint2*)&sA[0][amRow + 32][akq * 4] = u1;
  }
  asm volatile("s_wait_asynccnt 0x0" ::: "memory");
  __syncthreads();

  for (int i = 0; i < nkLoc; ++i) {
    const int cur = i & 1;
    const bool more = (i + 1) < nkLoc;
    uint2 n0, n1;
    if (more) {
      stageA(kt0 + i + 1, n0, n1);       // global gathers for next tile (regs)
      asyncB(kt0 + i + 1, cur ^ 1);      // async global->LDS for next B tile
      if (i + 2 < nkLoc)
        __builtin_prefetch((const void*)(wrow + ((kt0 + i + 2) << 5)), 0, 3);
    }
    // ---- compute current tile: 12 ds_load_b128, 8 WMMA
    union F { v16h v; v8h h[2]; } fa[2];
#pragma unroll
    for (int mi = 0; mi < 2; ++mi) {
      int row = mi * 32 + wm * 16 + lrow;
      fa[mi].h[0] = *(const v8h*)&sA[cur][row][half * 8];
      fa[mi].h[1] = *(const v8h*)&sA[cur][row][16 + half * 8];
    }
#pragma unroll
    for (int t = 0; t < 4; ++t) {
      F fb;
      int nl = wn * 64 + t * 16 + lrow;
      fb.h[0] = *(const v8h*)&sB[cur][nl][half * 8];
      fb.h[1] = *(const v8h*)&sB[cur][nl][16 + half * 8];
#pragma unroll
      for (int mi = 0; mi < 2; ++mi)
        acc[mi][t] = __builtin_amdgcn_wmma_f32_16x16x32_f16(
            false, fa[mi].v, false, fb.v, (short)0, acc[mi][t], false, false);
    }
    if (more) {
      *(uint2*)&sA[cur ^ 1][amRow][akq * 4] = n0;
      *(uint2*)&sA[cur ^ 1][amRow + 32][akq * 4] = n1;
    }
    asm volatile("s_wait_asynccnt 0x0" ::: "memory");
    __syncthreads();
  }

  // ---- epilogue: D layout -> N = lane&15, M = r + 8*(lane>=16)
#pragma unroll
  for (int mi = 0; mi < 2; ++mi)
#pragma unroll
    for (int t = 0; t < 4; ++t) {
      int nn = Nbase + wn * 64 + t * 16 + lrow;
      if (nn < Cout) {
        float bs = SPLITK ? (blockIdx.z == 0 ? bias[nn] : 0.f) : bias[nn];
#pragma unroll
        for (int r = 0; r < 8; ++r) {
          int mm = Mbase + mi * 32 + wm * 16 + half * 8 + r;
          int b2 = mm / Msz; int ms2 = mm - b2 * Msz;
          float v = acc[mi][t][r] + bs;
          if (RELU) v = fmaxf(v, 0.f);
          long oidx = ((long)b2 * Cout + nn) * Msz + ms2;
          if (SPLITK)      atomicAdd((float*)out + oidx, v);
          else if (OUTF16) ((_Float16*)out)[oidx] = (_Float16)v;
          else             ((float*)out)[oidx] = v;
        }
      }
    }
}

// ------------------------------- primary-capsule reshape + squash -> u (B,2048,8)
__global__ void squash_u_k(const float* __restrict__ p, float* __restrict__ u) {
  int idx = blockIdx.x * blockDim.x + threadIdx.x;
  if (idx >= BB * 2048) return;
  int b = idx >> 11, cap = idx & 2047;
  int g = cap >> 6, vsp = cap & 63;
  float s[8]; float sq = 0.f;
#pragma unroll
  for (int d = 0; d < 8; ++d) {
    s[d] = p[((long)b * 256 + g * 8 + d) * 64 + vsp];
    sq += s[d] * s[d];
  }
  float sc = (sq / (1.f + sq)) * rsqrtf(sq + 1e-8f);
#pragma unroll
  for (int d = 0; d < 8; ++d) u[(long)idx * 8 + d] = s[d] * sc;
}

// routing (n_iter=1 -> uniform c=1/50): v[b,j,:] = squash((1/50) sum_{i,c} W[j,i,:,c] u[b,i,c])
__global__ void routing_k(const float* __restrict__ u, const float* __restrict__ rw,
                          float* __restrict__ vout) {
  int bj = blockIdx.x; int b = bj / 50, j = bj % 50;
  int tid = threadIdx.x;
  int d = tid & 63, ig = tid >> 6;
  float acc = 0.f;
  for (int i = ig; i < 2048; i += 4) {
    const float* wr = rw + (((long)j * 2048 + i) * 64 + d) * 8;
    const float* ur = u + ((long)b * 2048 + i) * 8;
#pragma unroll
    for (int c2 = 0; c2 < 8; ++c2) acc += wr[c2] * ur[c2];
  }
  __shared__ float red[256];
  __shared__ float sv[64];
  __shared__ float sscale;
  red[tid] = acc;
  __syncthreads();
  if (tid < 64)
    sv[tid] = (red[tid] + red[tid + 64] + red[tid + 128] + red[tid + 192]) * 0.02f;
  __syncthreads();
  if (tid == 0) {
    float ss = 0.f;
    for (int k = 0; k < 64; ++k) ss += sv[k] * sv[k];
    sscale = (ss / (1.f + ss)) * rsqrtf(ss + 1e-8f);
  }
  __syncthreads();
  if (tid < 64) vout[((long)b * 50 + j) * 64 + tid] = sv[tid] * sscale;
}

// zero-dilate + asym-pad (f32 in -> f16 out) so ConvTranspose3d == plain conv
__global__ void dilate_pad_k(const float* __restrict__ in, _Float16* __restrict__ out,
                             int BC, int IS, int OS, int stride, int pl) {
  long total = (long)BC * OS * OS * OS;
  long idx = (long)blockIdx.x * blockDim.x + threadIdx.x;
  if (idx >= total) return;
  int x = idx % OS; long r = idx / OS;
  int y = r % OS; r /= OS;
  int z = r % OS; long bc = r / OS;
  float val = 0.f;
  int tz = z - pl, ty = y - pl, tx = x - pl;
  if (tz >= 0 && ty >= 0 && tx >= 0 &&
      tz % stride == 0 && ty % stride == 0 && tx % stride == 0) {
    int iz = tz / stride, iy = ty / stride, ix = tx / stride;
    if (iz < IS && iy < IS && ix < IS)
      val = in[((bc * IS + iz) * IS + iy) * IS + ix];
  }
  out[idx] = (_Float16)val;
}

// gather per-point features: out[b,p,co] = d2[b,co,vid]
__global__ void gather_k(const float* __restrict__ d2, const int* __restrict__ vid,
                         float* __restrict__ out, int Pn) {
  int bp = blockIdx.x; int b = bp / Pn, p = bp % Pn;
  int v = vid[(long)b * Pn + p];
  int c = threadIdx.x;
#pragma unroll
  for (int rr = 0; rr < 2; ++rr) {
    int co = c + rr * 256;
    out[(long)bp * 512 + co] = d2[((long)b * 512 + co) * NVOX3 + v];
  }
}

// -------------------------------------------------------------------- launch
extern "C" void kernel_launch(void* const* d_in, const int* in_sizes, int n_in,
                              void* d_out, int out_size, void* d_ws, size_t ws_size,
                              hipStream_t stream) {
  const float* pcl  = (const float*)d_in[0];
  const float* feat = (const float*)d_in[1];
  // d_in[2] = n (== 24, hardcoded as NVOX)
  const float* c1w = (const float*)d_in[3];
  const float* c1b = (const float*)d_in[4];
  const float* c2w = (const float*)d_in[5];
  const float* c2b = (const float*)d_in[6];
  const float* pw  = (const float*)d_in[7];
  const float* pb  = (const float*)d_in[8];
  const float* rw  = (const float*)d_in[9];
  const float* d1w = (const float*)d_in[10];
  const float* d1b = (const float*)d_in[11];
  const float* d2w = (const float*)d_in[12];
  const float* d2b = (const float*)d_in[13];
  (void)in_sizes; (void)n_in; (void)out_size; (void)ws_size;

  char* ws = (char*)d_ws;
  size_t off = 0;
  auto alloc = [&](size_t nbytes) -> void* {
    void* p = ws + off;
    off += (nbytes + 255) & ~(size_t)255;
    return p;
  };
  float*    mnmx  = (float*)   alloc((size_t)BB * 6 * sizeof(float));
  int*      vid   = (int*)     alloc((size_t)BB * PP * sizeof(int));
  float*    mesh  = (float*)   alloc((size_t)BB * 128 * NVOX3 * sizeof(float));
  _Float16* meshh = (_Float16*)alloc((size_t)BB * 128 * NVOX3 * sizeof(_Float16));
  _Float16* w1h   = (_Float16*)alloc((size_t)256 * 16000 * sizeof(_Float16));
  _Float16* c1oh  = (_Float16*)alloc((size_t)BB * 256 * 8000 * sizeof(_Float16));
  _Float16* w2h   = (_Float16*)alloc((size_t)256 * 32000 * sizeof(_Float16));
  _Float16* c2oh  = (_Float16*)alloc((size_t)BB * 256 * 4096 * sizeof(_Float16));
  _Float16* pwh   = (_Float16*)alloc((size_t)256 * 186624 * sizeof(_Float16));
  float*    po    = (float*)   alloc((size_t)BB * 256 * 64 * sizeof(float));
  float*    u     = (float*)   alloc((size_t)BB * 2048 * 8 * sizeof(float));
  float*    vcap  = (float*)   alloc((size_t)BB * 50 * 64 * sizeof(float));
  _Float16* d1pad = (_Float16*)alloc((size_t)BB * 50 * 14 * 14 * 14 * sizeof(_Float16));
  _Float16* w1ph  = (_Float16*)alloc((size_t)256 * 1376 * sizeof(_Float16));
  float*    d1o   = (float*)   alloc((size_t)BB * 128 * 12 * 12 * 12 * sizeof(float));
  _Float16* d2pad = (_Float16*)alloc((size_t)BB * 128 * 26 * 26 * 26 * sizeof(_Float16));
  _Float16* w2ph  = (_Float16*)alloc((size_t)512 * 3456 * sizeof(_Float16));
  float*    d2o   = (float*)   alloc((size_t)BB * 512 * NVOX3 * sizeof(float));
  int*      off1  = (int*)     alloc((size_t)16000 * sizeof(int));
  int*      off2  = (int*)     alloc((size_t)32000 * sizeof(int));
  int*      offp  = (int*)     alloc((size_t)186624 * sizeof(int));
  int*      offd1 = (int*)     alloc((size_t)1376 * sizeof(int));
  int*      offd2 = (int*)     alloc((size_t)3456 * sizeof(int));

  // voxelize
  zero_f32<<<1024, 256, 0, stream>>>(mesh, (long)BB * 128 * NVOX3);
  minmax_k<<<BB, 256, 0, stream>>>(pcl, mnmx, PP);
  scatter_k<<<BB * PP, 128, 0, stream>>>(pcl, feat, mnmx, mesh, vid, PP, 128);
  cvt_f16_k<<<2048, 256, 0, stream>>>(mesh, meshh, (long)BB * 128 * NVOX3);

  // weight + offset-table prep
  {
    long t1 = (long)256 * 16000;
    prep_wdir_f16<<<(int)((t1 + 255) / 256), 256, 0, stream>>>(c1w, w1h, 256, 16000, 16000, t1);
    long t2 = (long)256 * 32000;
    prep_wdir_f16<<<(int)((t2 + 255) / 256), 256, 0, stream>>>(c2w, w2h, 256, 32000, 32000, t2);
    long t3 = (long)256 * 186624;
    prep_wdir_f16<<<(int)((t3 + 255) / 256), 256, 0, stream>>>(pw, pwh, 256, 186624, 186624, t3);
    long t4 = (long)256 * 1376;
    prep_wdec_f16<<<(int)((t4 + 255) / 256), 256, 0, stream>>>(d1w, w1ph, 50, 128, 1376, t4);
    long t5 = (long)512 * 3456;
    prep_wdec_f16<<<(int)((t5 + 255) / 256), 256, 0, stream>>>(d2w, w2ph, 128, 512, 3456, t5);
    prep_offs_k<<<(16000 + 255) / 256, 256, 0, stream>>>(off1, 128, 5, 576, 24, 13824, 16000, 16000);
    prep_offs_k<<<(32000 + 255) / 256, 256, 0, stream>>>(off2, 256, 5, 400, 20, 8000, 32000, 32000);
    prep_offs_k<<<(186624 + 255) / 256, 256, 0, stream>>>(offp, 256, 9, 256, 16, 4096, 186624, 186624);
    prep_offs_k<<<(1376 + 255) / 256, 256, 0, stream>>>(offd1, 50, 3, 196, 14, 2744, 1350, 1376);
    prep_offs_k<<<(3456 + 255) / 256, 256, 0, stream>>>(offd2, 128, 3, 676, 26, 17576, 3456, 3456);
  }

  dim3 blk(256);
  // encoder: 24 -> 20 -> 16 (relu, f16 out)
  conv3d_wmma<true, true, false><<<dim3(250, 1, 1), blk, 0, stream>>>(
      meshh, w1h, c1b, off1, c1oh, 128, 256, 1, 24, 24, 24, 20, 20, 20, 16000);
  conv3d_wmma<true, true, false><<<dim3(128, 1, 1), blk, 0, stream>>>(
      c1oh, w2h, c2b, off2, c2oh, 256, 256, 1, 20, 20, 20, 16, 16, 16, 32000);
  // primary caps k=9 s=2 -> 4 (no act, f32 out, split-K 24-way with atomic acc)
  zero_f32<<<64, 256, 0, stream>>>(po, (long)BB * 256 * 64);
  conv3d_wmma<false, false, true><<<dim3(2, 1, 24), blk, 0, stream>>>(
      c2oh, pwh, pb, offp, po, 256, 256, 2, 16, 16, 16, 4, 4, 4, 186624);

  squash_u_k<<<(BB * 2048 + 255) / 256, 256, 0, stream>>>(po, u);
  routing_k<<<BB * 50, 256, 0, stream>>>(u, rw, vcap);

  // dec1: ConvT(50->128,k3,s3) == conv over dilated(3)+pad(2) 4->14, out 12
  dilate_pad_k<<<(int)(((long)BB * 50 * 14 * 14 * 14 + 255) / 256), 256, 0, stream>>>(
      vcap, d1pad, BB * 50, 4, 14, 3, 2);
  conv3d_wmma<true, false, false><<<dim3(54, 1, 1), blk, 0, stream>>>(
      d1pad, w1ph, d1b, offd1, d1o, 50, 128, 1, 14, 14, 14, 12, 12, 12, 1350);

  // dec2: ConvT(128->512,k3,s2,p1,op1) == conv over dilated(2)+pad(1) 12->26, out 24
  dilate_pad_k<<<(int)(((long)BB * 128 * 26 * 26 * 26 + 255) / 256), 256, 0, stream>>>(
      d1o, d2pad, BB * 128, 12, 26, 2, 1);
  conv3d_wmma<true, false, false><<<dim3(432, 2, 1), blk, 0, stream>>>(
      d2pad, w2ph, d2b, offd2, d2o, 128, 512, 1, 26, 26, 26, 24, 24, 24, 3456);

  gather_k<<<BB * PP, 256, 0, stream>>>(d2o, vid, (float*)d_out, PP);
}
